// Critic_17746804867899
// MI455X (gfx1250) — matrix operand order repaired and checked by reading, not compiled, not run on hardware
//
#include <hip/hip_runtime.h>
#include <hip/hip_bf16.h>

// ---------------------------------------------------------------------------
// LayerNorm-LSTM critic, fully fused for gfx1250 (MI455X), 2 barriers/step.
//   Waves 0-3: one full gate each -> 24 x v_wmma_f32_16x16x32_bf16 per step.
//     Wh fragments (recurrent path) + q=0 Wx fragments held in registers so
//     WMMAs issue without waiting on ds loads; gate-LN done in registers via
//     half-wave shfl_xor over the WMMA C layout.
//   Waves 4-7: stage x(t+1) (fp32->bf16) into double-buffered LDS tile.
//   All waves: fused cell update + cell-LN + h + q-dot with b128 LDS accesses.
// ---------------------------------------------------------------------------

typedef __attribute__((ext_vector_type(16))) __bf16 v16bf;
typedef __attribute__((ext_vector_type(8)))  __bf16 v8bf;
typedef __attribute__((ext_vector_type(4)))  __bf16 v4bf;
typedef __attribute__((ext_vector_type(8)))  float  v8f;

#define N_SEQ 2048
#define T_STEPS 256
#define OBS 96
#define ACT 32
#define HID 64
#define DIN 128           // OBS+ACT
#define G4  256           // 4*HID

// LDS carve (bytes):
//  z    : 16*260 f32 = 16640   (normalized gates, padded stride)
//  cst  : 16*64  f32 =  4096   (cell state, post-LN)
//  gam  : 5*64   f32 =  1280
//  bet  : 5*64   f32 =  1280
//  dw   : 72     f32 =   288   (dw[64] = dense_b)
//  bx   : 32768 bf16 = 65536   (packed Wx B-fragments, q=0..3, c=0..15)
//  bh   : 16384 bf16 = 32768   (packed Wh B-fragments, q=0..1)
//  xA0/1: 16*136 bf16 = 4352*2 (double-buffered x tile)
//  hA   : 16*72  bf16 =  2304
#define SMEM_BYTES (16640 + 4096 + 1280 + 1280 + 288 + 65536 + 32768 + 4352*2 + 2304)

__device__ __forceinline__ float fexp(float x) {
    return __builtin_amdgcn_exp2f(x * 1.4426950408889634f);   // v_exp_f32
}
__device__ __forceinline__ float sigm(float x) {
    return __builtin_amdgcn_rcpf(1.0f + fexp(-x));            // v_rcp_f32
}
__device__ __forceinline__ float ftanh(float x) {
    return 2.0f * sigm(2.0f * x) - 1.0f;
}

// -------------------- prep: pack W into WMMA B-fragment layout --------------
// B (32x16 bf16) in contiguous k order (matches load_afrag): lanes 0-15 -> K=0..15 (N=lane), lanes 16-31
// -> K=16..31 (N=lane-16); half h of the 16-half record = K=kgroup+h.
__global__ void pack_w_kernel(const float* __restrict__ W,
                              __bf16* __restrict__ bx,
                              __bf16* __restrict__ bh) {
    int idx = blockIdx.x * blockDim.x + threadIdx.x;   // 0..3071
    if (idx < 2048) {                                  // Wx: 64 frags * 32 lanes
        int lane = idx & 31, frag = idx >> 5;
        int q = frag >> 4, c = frag & 15;
        int krow = 32 * q + ((lane & 16) ? 16 : 0);
        int col  = 16 * c + (lane & 15);
        __bf16* dst = bx + (size_t)idx * 16;
#pragma unroll
        for (int h = 0; h < 16; ++h) dst[h] = (__bf16)W[(krow + h) * G4 + col];
    } else if (idx < 3072) {                           // Wh: 32 frags * 32 lanes
        int j = idx - 2048;
        int lane = j & 31, frag = j >> 5;
        int q = frag >> 4, c = frag & 15;
        int krow = DIN + 32 * q + ((lane & 16) ? 16 : 0);
        int col  = 16 * c + (lane & 15);
        __bf16* dst = bh + (size_t)j * 16;
#pragma unroll
        for (int h = 0; h < 16; ++h) dst[h] = (__bf16)W[(krow + h) * G4 + col];
    }
}

// A fragment (16x32 bf16): lane L -> row M=L&15; lanes 0-15
// read K = 32q+[0..15]; lanes 16-31 read K = 32q+16+[0..15].
__device__ __forceinline__ v16bf load_afrag(const __bf16* base, int stride,
                                            int q, int lane) {
    int M  = lane & 15;
    int kb = (lane & 16) ? 16 : 0;
    const __bf16* p = base + M * stride + 32 * q + kb;
    v8bf lo = *(const v8bf*)(p);
    v8bf hi = *(const v8bf*)(p + 8);
    return __builtin_shufflevector(lo, hi, 0,1,2,3,4,5,6,7,8,9,10,11,12,13,14,15);
}

__device__ __forceinline__ v16bf load_bfrag(const __bf16* arr, int frag, int lane) {
    return *(const v16bf*)(arr + ((size_t)frag * 32 + lane) * 16);
}

// stage one 16x128 x_t tile (fp32 -> bf16) with 128 threads (16 floats each)
__device__ __forceinline__ void stage_x(const float* __restrict__ obss,
                                        const float* __restrict__ actions,
                                        int n0, int t, __bf16* buf, int sid) {
    int row = sid >> 3;
    int k0  = (sid & 7) * 16;
    const float* src = (k0 < OBS)
        ? obss    + ((size_t)(n0 + row) * T_STEPS + t) * OBS + k0
        : actions + ((size_t)(n0 + row) * T_STEPS + t) * ACT + (k0 - OBS);
    float4 a0 = ((const float4*)src)[0];
    float4 a1 = ((const float4*)src)[1];
    float4 a2 = ((const float4*)src)[2];
    float4 a3 = ((const float4*)src)[3];
    v8bf v0, v1;
    v0[0]=(__bf16)a0.x; v0[1]=(__bf16)a0.y; v0[2]=(__bf16)a0.z; v0[3]=(__bf16)a0.w;
    v0[4]=(__bf16)a1.x; v0[5]=(__bf16)a1.y; v0[6]=(__bf16)a1.z; v0[7]=(__bf16)a1.w;
    v1[0]=(__bf16)a2.x; v1[1]=(__bf16)a2.y; v1[2]=(__bf16)a2.z; v1[3]=(__bf16)a2.w;
    v1[4]=(__bf16)a3.x; v1[5]=(__bf16)a3.y; v1[6]=(__bf16)a3.z; v1[7]=(__bf16)a3.w;
    *(v8bf*)(buf + row * 136 + k0)     = v0;
    *(v8bf*)(buf + row * 136 + k0 + 8) = v1;
    if (t + 1 < T_STEPS)                 // WGP-scope prefetch of next step
        __builtin_prefetch(src + ((k0 < OBS) ? OBS : ACT), 0, 3);
}

__global__ void __launch_bounds__(256, 1)
lnlstm_kernel(const float* __restrict__ obss,
              const float* __restrict__ actions,
              const __bf16* __restrict__ bx_ws,
              const __bf16* __restrict__ bh_ws,
              const float* __restrict__ ln_gamma,
              const float* __restrict__ ln_beta,
              const float* __restrict__ dense_W,
              const float* __restrict__ dense_b,
              float* __restrict__ out) {
    extern __shared__ char smem[];
    float*  z   = (float*)smem;                 // [16][260]
    float*  cst = z   + 16 * 260;               // [16][64]
    float*  gam = cst + 16 * 64;                // [5][64]
    float*  bet = gam + 5 * 64;                 // [5][64]
    float*  dw  = bet + 5 * 64;                 // [72]
    __bf16* bxl = (__bf16*)(dw + 72);           // 32768
    __bf16* bhl = bxl + 32768;                  // 16384
    __bf16* xA0 = bhl + 16384;                  // [16][136]
    __bf16* xA1 = xA0 + 16 * 136;               // [16][136]
    __bf16* hA  = xA1 + 16 * 136;               // [16][72]

    const int tid  = threadIdx.x;
    const int lane = tid & 31;
    const int wid  = tid >> 5;                  // 8 waves
    const int n0   = blockIdx.x * 16;           // 16 sequences per WG

    // ---- one-time init ----
    for (int i = tid; i < 6144; i += 256)       // bx+bh contiguous: 96KB
        ((float4*)bxl)[i] = ((const float4*)bx_ws)[i];
    for (int i = tid; i < 1024; i += 256) cst[i] = 0.0f;
    for (int i = tid; i < 16 * 72; i += 256) hA[i] = (__bf16)0.0f;
    for (int i = tid; i < 320; i += 256) { gam[i] = ln_gamma[i]; bet[i] = ln_beta[i]; }
    if (tid < 64) dw[tid] = dense_W[tid];
    if (tid == 0) dw[64] = dense_b[0];
    if (tid < 128) stage_x(obss, actions, n0, 0, xA0, tid);
    __syncthreads();

    // ---- hoisted per-thread constants ----
    const int cn = lane & 15;                   // WMMA C-layout column in tile
    const int mb = (lane & 16) ? 8 : 0;         // WMMA C-layout row base
    float gv0=0,gv1=0,gv2=0,gv3=0, bv0=0,bv1=0,bv2=0,bv3=0;
    // loop-invariant B fragments held in registers:
    //   all 8 Wh fragments (recurrent critical path) + the 4 q=0 Wx fragments
    v16bf hb00={}, hb01={}, hb02={}, hb03={};
    v16bf hb10={}, hb11={}, hb12={}, hb13={};
    v16bf xb00={}, xb01={}, xb02={}, xb03={};
    if (wid < 4) {                              // gate LN gamma/beta, per lane
        const int g = wid;
        gv0 = gam[g*64 +  0 + cn]; bv0 = bet[g*64 +  0 + cn];
        gv1 = gam[g*64 + 16 + cn]; bv1 = bet[g*64 + 16 + cn];
        gv2 = gam[g*64 + 32 + cn]; bv2 = bet[g*64 + 32 + cn];
        gv3 = gam[g*64 + 48 + cn]; bv3 = bet[g*64 + 48 + cn];
        hb00 = load_bfrag(bhl,      g*4 + 0, lane);
        hb01 = load_bfrag(bhl,      g*4 + 1, lane);
        hb02 = load_bfrag(bhl,      g*4 + 2, lane);
        hb03 = load_bfrag(bhl,      g*4 + 3, lane);
        hb10 = load_bfrag(bhl, 16 + g*4 + 0, lane);
        hb11 = load_bfrag(bhl, 16 + g*4 + 1, lane);
        hb12 = load_bfrag(bhl, 16 + g*4 + 2, lane);
        hb13 = load_bfrag(bhl, 16 + g*4 + 3, lane);
        xb00 = load_bfrag(bxl,      g*4 + 0, lane);
        xb01 = load_bfrag(bxl,      g*4 + 1, lane);
        xb02 = load_bfrag(bxl,      g*4 + 2, lane);
        xb03 = load_bfrag(bxl,      g*4 + 3, lane);
    }
    const int frow = tid >> 4;                  // fused block: row per 16 lanes
    const int e0   = (tid & 15) * 4;            // 4 elements per thread
    float g4k[4], b4k[4], dwk[4];
#pragma unroll
    for (int k = 0; k < 4; ++k) {
        g4k[k] = gam[4*64 + e0 + k];
        b4k[k] = bet[4*64 + e0 + k];
        dwk[k] = dw[e0 + k];
    }
    const float dbv = dw[64];

    for (int t = 0; t < T_STEPS; ++t) {
        const __bf16* xcur = (t & 1) ? xA1 : xA0;
        __bf16*       xnxt = (t & 1) ? xA0 : xA1;

        if (wid < 4) {
            // ---- one full gate per wave: 24 WMMAs ----
            const int g = wid;
            v8f acc0 = {}, acc1 = {}, acc2 = {}, acc3 = {};
            {   // q = 0 for x: B operands already in registers
                v16bf a = load_afrag(xcur, 136, 0, lane);
                acc0 = __builtin_amdgcn_wmma_f32_16x16x32_bf16(false, a, false, xb00, (short)0, acc0, false, false);
                acc1 = __builtin_amdgcn_wmma_f32_16x16x32_bf16(false, a, false, xb01, (short)0, acc1, false, false);
                acc2 = __builtin_amdgcn_wmma_f32_16x16x32_bf16(false, a, false, xb02, (short)0, acc2, false, false);
                acc3 = __builtin_amdgcn_wmma_f32_16x16x32_bf16(false, a, false, xb03, (short)0, acc3, false, false);
            }
#pragma unroll
            for (int q = 1; q < 4; ++q) {
                v16bf a = load_afrag(xcur, 136, q, lane);
                acc0 = __builtin_amdgcn_wmma_f32_16x16x32_bf16(false, a, false,
                        load_bfrag(bxl, q*16 + g*4 + 0, lane), (short)0, acc0, false, false);
                acc1 = __builtin_amdgcn_wmma_f32_16x16x32_bf16(false, a, false,
                        load_bfrag(bxl, q*16 + g*4 + 1, lane), (short)0, acc1, false, false);
                acc2 = __builtin_amdgcn_wmma_f32_16x16x32_bf16(false, a, false,
                        load_bfrag(bxl, q*16 + g*4 + 2, lane), (short)0, acc2, false, false);
                acc3 = __builtin_amdgcn_wmma_f32_16x16x32_bf16(false, a, false,
                        load_bfrag(bxl, q*16 + g*4 + 3, lane), (short)0, acc3, false, false);
            }
            {   // h part: all B operands in registers, only A comes from LDS
                v16bf a = load_afrag(hA, 72, 0, lane);
                acc0 = __builtin_amdgcn_wmma_f32_16x16x32_bf16(false, a, false, hb00, (short)0, acc0, false, false);
                acc1 = __builtin_amdgcn_wmma_f32_16x16x32_bf16(false, a, false, hb01, (short)0, acc1, false, false);
                acc2 = __builtin_amdgcn_wmma_f32_16x16x32_bf16(false, a, false, hb02, (short)0, acc2, false, false);
                acc3 = __builtin_amdgcn_wmma_f32_16x16x32_bf16(false, a, false, hb03, (short)0, acc3, false, false);
                v16bf a2 = load_afrag(hA, 72, 1, lane);
                acc0 = __builtin_amdgcn_wmma_f32_16x16x32_bf16(false, a2, false, hb10, (short)0, acc0, false, false);
                acc1 = __builtin_amdgcn_wmma_f32_16x16x32_bf16(false, a2, false, hb11, (short)0, acc1, false, false);
                acc2 = __builtin_amdgcn_wmma_f32_16x16x32_bf16(false, a2, false, hb12, (short)0, acc2, false, false);
                acc3 = __builtin_amdgcn_wmma_f32_16x16x32_bf16(false, a2, false, hb13, (short)0, acc3, false, false);
            }
            // ---- in-register gate LN: row m = mb+r lives in 16 lanes x 4 accs
#pragma unroll
            for (int r = 0; r < 8; ++r) {
                float s  = acc0[r] + acc1[r] + acc2[r] + acc3[r];
                float s2 = acc0[r]*acc0[r] + acc1[r]*acc1[r]
                         + acc2[r]*acc2[r] + acc3[r]*acc3[r];
                s  += __shfl_xor(s, 1);  s  += __shfl_xor(s, 2);
                s  += __shfl_xor(s, 4);  s  += __shfl_xor(s, 8);
                s2 += __shfl_xor(s2, 1); s2 += __shfl_xor(s2, 2);
                s2 += __shfl_xor(s2, 4); s2 += __shfl_xor(s2, 8);
                float mean = s * (1.0f / 64.0f);
                float var  = s2 * (1.0f / 64.0f) - mean * mean;
                float rs   = __builtin_amdgcn_rsqf(var + 1e-12f);   // v_rsq_f32
                float* zr = z + (mb + r) * 260 + g * 64 + cn;
                zr[ 0] = (acc0[r] - mean) * rs * gv0 + bv0;
                zr[16] = (acc1[r] - mean) * rs * gv1 + bv1;
                zr[32] = (acc2[r] - mean) * rs * gv2 + bv2;
                zr[48] = (acc3[r] - mean) * rs * gv3 + bv3;
            }
        } else {
            // ---- waves 4-7: stage x(t+1) into the other buffer ----
            if (t + 1 < T_STEPS)
                stage_x(obss, actions, n0, t + 1, xnxt, tid - 128);
        }
        __syncthreads();

        // ---- fused cell update + cell-LN + h + q-dot (b128 LDS accesses) ----
        const float* zrow = z + frow * 260;
        float4 iv = *(const float4*)(zrow + e0);
        float4 jv = *(const float4*)(zrow + 64 + e0);
        float4 fv = *(const float4*)(zrow + 128 + e0);
        float4 ovv = *(const float4*)(zrow + 192 + e0);
        float4 cv = *(const float4*)(cst + frow * 64 + e0);
        float ia[4] = {iv.x, iv.y, iv.z, iv.w};
        float ja[4] = {jv.x, jv.y, jv.z, jv.w};
        float fa[4] = {fv.x, fv.y, fv.z, fv.w};
        float oa[4] = {ovv.x, ovv.y, ovv.z, ovv.w};
        float ca[4] = {cv.x, cv.y, cv.z, cv.w};
        float nc[4];
        float s = 0.f, s2 = 0.f;
#pragma unroll
        for (int k = 0; k < 4; ++k) {
            float v = ca[k] * sigm(fa[k] + 1.0f) + sigm(ia[k]) * ftanh(ja[k]);
            nc[k] = v; s += v; s2 += v * v;
        }
        s  += __shfl_xor(s, 1);  s  += __shfl_xor(s, 2);
        s  += __shfl_xor(s, 4);  s  += __shfl_xor(s, 8);
        s2 += __shfl_xor(s2, 1); s2 += __shfl_xor(s2, 2);
        s2 += __shfl_xor(s2, 4); s2 += __shfl_xor(s2, 8);
        float mean = s * (1.0f / 64.0f);
        float var  = s2 * (1.0f / 64.0f) - mean * mean;
        float rs   = __builtin_amdgcn_rsqf(var + 1e-12f);
        float qp = 0.f;
        float4 cno4;
        v4bf hb;
#pragma unroll
        for (int k = 0; k < 4; ++k) {
            float cno = (nc[k] - mean) * rs * g4k[k] + b4k[k];
            ((float*)&cno4)[k] = cno;
            float h = ftanh(cno) * sigm(oa[k]);
            hb[k] = (__bf16)h;
            qp += h * dwk[k];
        }
        *(float4*)(cst + frow * 64 + e0) = cno4;
        *(v4bf*)(hA + frow * 72 + e0)    = hb;
        qp += __shfl_xor(qp, 1); qp += __shfl_xor(qp, 2);
        qp += __shfl_xor(qp, 4); qp += __shfl_xor(qp, 8);
        if ((tid & 15) == 0)
            out[(size_t)(n0 + frow) * T_STEPS + t] = qp + dbv;
        __syncthreads();   // hA/cst ready before next step's WMMA/fused reads
    }
}

extern "C" void kernel_launch(void* const* d_in, const int* in_sizes, int n_in,
                              void* d_out, int out_size, void* d_ws, size_t ws_size,
                              hipStream_t stream) {
    const float* obss     = (const float*)d_in[0];
    const float* actions  = (const float*)d_in[1];
    const float* W        = (const float*)d_in[2];
    const float* ln_gamma = (const float*)d_in[3];
    const float* ln_beta  = (const float*)d_in[4];
    const float* dense_W  = (const float*)d_in[5];
    const float* dense_b  = (const float*)d_in[6];
    float* out = (float*)d_out;

    __bf16* bx_ws = (__bf16*)d_ws;            // 64KB of packed Wx fragments
    __bf16* bh_ws = bx_ws + 32768;            // 32KB of packed Wh fragments

    pack_w_kernel<<<12, 256, 0, stream>>>(W, bx_ws, bh_ws);
    lnlstm_kernel<<<N_SEQ / 16, 256, SMEM_BYTES, stream>>>(
        obss, actions, bx_ws, bh_ws, ln_gamma, ln_beta, dense_W, dense_b, out);
}
